// IEGMNetXNOR_283467841701
// MI455X (gfx1250) — 8-bit matrix operand order repaired and checked by reading, compiled, not run on hardware
//
#include <hip/hip_runtime.h>


typedef __attribute__((ext_vector_type(8))) int   v8i;
typedef __attribute__((ext_vector_type(8))) float v8f;

#define B_TOTAL 16384

__device__ __forceinline__ float q8f(float x, float lo) {
  x = fminf(fmaxf(x, lo), 1.0f);
  return rintf(x * 8.0f) * 0.125f;
}

__device__ __forceinline__ unsigned char fp8_e4m3(float v) {
  union { float f; unsigned u; } c; c.f = v;
  if (c.f == 0.0f) return 0;
  unsigned s = c.u >> 31;
  int e = (int)((c.u >> 23) & 0xffu) - 127 + 7;
  unsigned m = (c.u >> 20) & 7u;
  return (unsigned char)((s << 7) | ((unsigned)e << 3) | m);
}

__global__ void __launch_bounds__(256)
quant_to_fp8_kernel(const float* __restrict__ x, unsigned char* __restrict__ q, long n) {
  long i  = (long)blockIdx.x * blockDim.x + threadIdx.x;
  long st = (long)gridDim.x * blockDim.x;
  for (; i < n; i += st) q[i] = fp8_e4m3(q8f(x[i], -1.0f));
}

__global__ void __launch_bounds__(256)
gemm_fp8_wmma_kernel(const unsigned char* __restrict__ in,
                     unsigned char* __restrict__ out8,
                     float* __restrict__ outf,
                     const float* __restrict__ w,
                     const float* __restrict__ bn_g, const float* __restrict__ bn_b,
                     const float* __restrict__ bn_m, const float* __restrict__ bn_v,
                     int InS, int OutS,
                     int Ci, int Co, int Kh, int Tout, int P,
                     int KP, int NC, int nBlocks, int tBlocks,
                     int out_t_stride, int out_c_stride, float clip_lo) {
  const int Keff = Ci * (2 * (P - 1) + Kh);

  __shared__ alignas(16) unsigned char wlds[12288];
  const int wtotal = nBlocks * 16 * KP;
  for (int i = threadIdx.x; i < wtotal; i += blockDim.x) {
    int cg = i / KP, k = i % KP;
    unsigned char byte = 0;
    if (cg < NC && k < Keff) {
      int dlt = cg / Co, co = cg % Co;
      int jp = k / Ci, ci = k % Ci;
      int j = jp - 2 * dlt;
      if (j >= 0 && j < Kh)
        byte = fp8_e4m3(q8f(w[(co * Ci + ci) * Kh + j], -1.0f));
    }
    wlds[i] = byte;
  }
  __syncthreads();

  const int lane = threadIdx.x & 31;
  const int n16  = lane & 15;
  const int half = lane >> 4;
  const int wid    = (int)((blockIdx.x * blockDim.x + threadIdx.x) >> 5);
  const int nWaves = (int)((gridDim.x * blockDim.x) >> 5);
  const int kSteps = KP >> 6;
  const int tasks  = (B_TOTAL / 16) * tBlocks;

  for (int task = wid; task < tasks; task += nWaves) {
    int tb = task % tBlocks;
    int bt = task / tBlocks;
    int b0 = bt << 4;
    int tbase = tb * P;
    const unsigned char* arow =
        in + (size_t)(b0 + n16) * InS + (size_t)(2 * tbase * Ci);

    int nt = task + nWaves;
    if (nt < tasks) {
      __builtin_prefetch(in + (size_t)(((nt / tBlocks) << 4) + n16) * InS +
                             (size_t)(2 * (nt % tBlocks) * P * Ci), 0, 1);
    }

    for (int nb = 0; nb < nBlocks; ++nb) {
      v8f acc = {};
      const unsigned char* wrow = wlds + (nb * 16 + n16) * KP + (half << 4);
      for (int ks = 0; ks < kSteps; ++ks) {
        v8i a, bf;
#pragma unroll
        for (int p = 0; p < 4; ++p) {
          unsigned long long d;
          __builtin_memcpy(&d, arow + (ks << 6) + ((p >> 1) << 5) + (half << 4) + ((p & 1) << 3), 8);
          a[2 * p]     = (int)(unsigned)d;
          a[2 * p + 1] = (int)(unsigned)(d >> 32);
        }
        const int4* wq = (const int4*)(wrow + (ks << 6));
        int4 blo = wq[0];
        int4 bhi = wq[2];
        bf[0] = blo.x; bf[1] = blo.y; bf[2] = blo.z; bf[3] = blo.w;
        bf[4] = bhi.x; bf[5] = bhi.y; bf[6] = bhi.z; bf[7] = bhi.w;
        acc = __builtin_amdgcn_wmma_f32_16x16x64_fp8_fp8(a, bf, (short)0, acc,
                                                         false, false);
      }
      int cg = nb * 16 + n16;
      if (cg < NC) {
        int dlt = cg / Co, co = cg % Co;
        int t = tbase + dlt;
        if (t < Tout) {
          float g  = bn_g[co], be = bn_b[co], mu = bn_m[co];
          float rv = rsqrtf(bn_v[co] + 1e-5f);
          size_t obase = (size_t)t * out_t_stride + (size_t)co * out_c_stride;
#pragma unroll
          for (int r = 0; r < 8; ++r) {
            int b = b0 + r + (half << 3);
            float y = g * (acc[r] - mu) * rv + be;
            if (outf) outf[(size_t)b * OutS + obase] = y;
            else      out8[(size_t)b * OutS + obase] = fp8_e4m3(q8f(y, clip_lo));
          }
        }
      }
    }
  }
}

static void launch_gemm(const unsigned char* in, unsigned char* out8, float* outf,
                        const float* w, const float* g, const float* b,
                        const float* m, const float* v,
                        int InS, int OutS, int Ci, int Co, int Kh, int Tout, int P,
                        int out_t_stride, int out_c_stride, float clip_lo,
                        hipStream_t stream) {
  int Keff = Ci * (2 * (P - 1) + Kh);
  int KP = ((Keff + 63) / 64) * 64;
  int NC = P * Co;
  int nBlocks = (NC + 15) / 16;
  int tBlocks = (Tout + P - 1) / P;
  long tasks = (long)(B_TOTAL / 16) * tBlocks;
  int blocks = (int)((tasks + 7) / 8);
  if (blocks > 2048) blocks = 2048;
  if (blocks < 1) blocks = 1;
  hipLaunchKernelGGL(gemm_fp8_wmma_kernel, dim3(blocks), dim3(256), 0, stream,
                     in, out8, outf, w, g, b, m, v,
                     InS, OutS, Ci, Co, Kh, Tout, P, KP, NC, nBlocks, tBlocks,
                     out_t_stride, out_c_stride, clip_lo);
}

extern "C" void kernel_launch(void* const* d_in, const int* in_sizes, int n_in,
                              void* d_out, int out_size, void* d_ws, size_t ws_size,
                              hipStream_t stream) {
  (void)in_sizes; (void)n_in; (void)out_size; (void)ws_size;
  const float* x    = (const float*)d_in[0];
  const float* w1   = (const float*)d_in[1];
  const float* w2   = (const float*)d_in[2];
  const float* w3   = (const float*)d_in[3];
  const float* w4   = (const float*)d_in[4];
  const float* w5   = (const float*)d_in[5];
  const float* fcw1 = (const float*)d_in[6];
  const float* fcw2 = (const float*)d_in[7];
#define BN(i) (const float*)d_in[(i)], (const float*)d_in[(i)+1], \
              (const float*)d_in[(i)+2], (const float*)d_in[(i)+3]

  unsigned char* bufA = (unsigned char*)d_ws;
  unsigned char* bufB = bufA + ((size_t)32 << 20);

  hipLaunchKernelGGL(quant_to_fp8_kernel, dim3(2048), dim3(256), 0, stream,
                     x, bufA, (long)B_TOTAL * 1250);

  launch_gemm(bufA, bufB, nullptr, w1, BN(8),  1250, 1869, 1, 3, 6, 623, 5, 3, 1, -1.0f, stream);
  launch_gemm(bufB, bufA, nullptr, w2, BN(12), 1869, 1550, 3, 5, 5, 310, 3, 5, 1, -1.0f, stream);
  launch_gemm(bufA, bufB, nullptr, w3, BN(16), 1550, 1540, 5, 10, 4, 154, 1, 10, 1, -1.0f, stream);
  launch_gemm(bufB, bufA, nullptr, w4, BN(20), 1540, 1520, 10, 20, 4, 76, 1, 20, 1, -1.0f, stream);
  launch_gemm(bufA, bufB, nullptr, w5, BN(24), 1520, 740, 20, 20, 4, 37, 1, 1, 37, -1.0f, stream);
  launch_gemm(bufB, bufA, nullptr, fcw1, BN(28), 740, 16, 740, 10, 1, 1, 1, 0, 1, 0.0f, stream);
  launch_gemm(bufA, nullptr, (float*)d_out, fcw2, BN(32), 16, 2, 10, 2, 1, 1, 1, 0, 1, 0.0f, stream);
#undef BN
}
